// InputFeedDecoder_35656818491732
// MI455X (gfx1250) — compile-verified, matrix operand order checked by reading, not run on hardware
//
#include <hip/hip_runtime.h>
#include <hip/hip_bf16.h>

// ---------------------------------------------------------------------------
// Problem constants (from reference): T=64, B=64, S=128, E=512, H=1024
// ---------------------------------------------------------------------------
constexpr int T = 64;
constexpr int B = 64;
constexpr int S = 128;
constexpr int E = 512;
constexpr int H = 1024;
constexpr int G4H = 4 * H;          // 4096 gate width
constexpr int KX = E + H;           // 1536 input-feed width
constexpr int K2 = 2 * H;           // 2048 out-projection K
constexpr int NT = 4;               // N-subtiles per wave in the GEMM

typedef __bf16 bf16;
typedef __attribute__((ext_vector_type(16))) __bf16 v16bf;
typedef __attribute__((ext_vector_type(8)))  __bf16 v8bf;
typedef __attribute__((ext_vector_type(8)))  float  v8f;

// ---------------------------------------------------------------------------
// WMMA bf16 GEMM:  C(MxN, f32) = A(MxK, bf16 row-major) * Bw(NxK, bf16)^T
// One wave per 16x64 C tile (NT=4 N-subtiles, 4 independent accumulators).
// K loop unrolled 2x with ping-pong fragment sets f0/f1: no rotation copies,
// stable disjoint register sets, progressive s_wait_loadcnt overlap of VMEM
// with the XDL pipe.  Requires K % 64 == 0, N % 64 == 0 (true for all shapes
// here: K in {1024,1536,2048}, N in {1024,4096}).  grid = (N/64, M/16).
//
// A-fragment layout (ISA 7.12.2, 16-bit A 16x32): row = lane&15,
//   K = (h>>3)*16 + (lane>>4)*8 + (h&7)  -> two contiguous 8-bf16 (16B) loads.
// B-fragment mirrors A with N <-> M (column = lane&15).
// C/D: VGPR r -> (M = r + 8*(lane>>4), N = lane&15).
// ---------------------------------------------------------------------------
struct Frag {
  v8bf a0, a1;
  v8bf b0[NT], b1[NT];
};

__device__ __forceinline__ void load_frag(Frag& f, const bf16* __restrict__ Arow,
                                          const bf16* __restrict__ Brow,
                                          long bstride, int k) {
  __builtin_prefetch(Arow + k + 64, 0, 1);       // global_prefetch_b8
  f.a0 = *(const v8bf*)(Arow + k);
  f.a1 = *(const v8bf*)(Arow + k + 16);
#pragma unroll
  for (int t = 0; t < NT; ++t) {
    const bf16* bp = Brow + t * bstride + k;
    __builtin_prefetch(bp + 64, 0, 1);
    f.b0[t] = *(const v8bf*)(bp);
    f.b1[t] = *(const v8bf*)(bp + 16);
  }
}

__device__ __forceinline__ void mma_frag(v8f (&acc)[NT], const Frag& f) {
  v16bf a;
#pragma unroll
  for (int i = 0; i < 8; ++i) { a[i] = f.a0[i]; a[i + 8] = f.a1[i]; }
#pragma unroll
  for (int t = 0; t < NT; ++t) {
    v16bf b;
#pragma unroll
    for (int i = 0; i < 8; ++i) { b[i] = f.b0[t][i]; b[i + 8] = f.b1[t][i]; }
    acc[t] = __builtin_amdgcn_wmma_f32_16x16x32_bf16(
        false, a, false, b, (short)0, acc[t], false, false);
  }
}

__launch_bounds__(32)
__global__ void gemm_wmma_bf16(const bf16* __restrict__ A, int lda,
                               const bf16* __restrict__ Bw, int ldb,
                               float* C, int ldc,
                               int K, const float* __restrict__ bias,
                               int accumulate)
{
  const int lane = threadIdx.x & 31;
  const int col0 = blockIdx.x * (16 * NT);
  const int row0 = blockIdx.y * 16;
  const int half = lane >> 4;       // 0 or 1 (selects K sub-group)
  const int l15  = lane & 15;

  const bf16* __restrict__ Arow = A  + (long)(row0 + l15) * lda + half * 8;
  const bf16* __restrict__ Brow = Bw + (long)(col0 + l15) * ldb + half * 8;
  const long bstride = (long)16 * ldb;    // 16 weight rows per N-subtile

  v8f acc[NT];
#pragma unroll
  for (int t = 0; t < NT; ++t) acc[t] = (v8f){0.f,0.f,0.f,0.f,0.f,0.f,0.f,0.f};

  Frag f0, f1;
  load_frag(f0, Arow, Brow, bstride, 0);

  int k0 = 0;
  for (; k0 + 64 < K; k0 += 64) {           // steady state: 2 K-steps/iter
    load_frag(f1, Arow, Brow, bstride, k0 + 32);
    mma_frag(acc, f0);
    load_frag(f0, Arow, Brow, bstride, k0 + 64);
    mma_frag(acc, f1);
  }
  // tail: K % 64 == 0 guaranteed -> exactly two K-steps remain
  load_frag(f1, Arow, Brow, bstride, k0 + 32);
  mma_frag(acc, f0);
  mma_frag(acc, f1);

  const int mbase = row0 + half * 8;
#pragma unroll
  for (int t = 0; t < NT; ++t) {
    const int n = col0 + t * 16 + l15;
    const float bv = bias ? bias[n] : 0.f;
#pragma unroll
    for (int r = 0; r < 8; ++r) {
      float* p = C + (long)(mbase + r) * ldc + n;
      float v = acc[t][r] + bv;
      if (accumulate) v += *p;
      *p = v;
    }
  }
}

// ---------------------------------------------------------------------------
// Setup / conversion kernels
// ---------------------------------------------------------------------------
__global__ void cvt_f32_bf16(const float* __restrict__ src, bf16* __restrict__ dst, int n) {
  int i = blockIdx.x * blockDim.x + threadIdx.x;
  if (i < n) dst[i] = (bf16)src[i];
}

// dst (C,R) = transpose of src (R,C), f32 -> bf16   (for W_a -> W_a^T)
__global__ void cvt_f32_bf16_T(const float* __restrict__ src, bf16* __restrict__ dst,
                               int R, int Cc) {
  int i = blockIdx.x * blockDim.x + threadIdx.x;
  if (i >= R * Cc) return;
  int r = i / Cc, c = i % Cc;
  dst[(long)c * R + r] = (bf16)src[i];
}

// context (S,B,H) f32 -> ctx_bt (B,S,H) bf16
__global__ void ctx_transpose_bf16(const float* __restrict__ src, bf16* __restrict__ dst) {
  long i = (long)blockIdx.x * blockDim.x + threadIdx.x;
  if (i >= (long)S * B * H) return;
  int s = (int)(i / ((long)B * H));
  int rem = (int)(i % ((long)B * H));
  int b = rem / H, h = rem % H;
  dst[((long)b * S + s) * H + h] = (bf16)src[i];
}

__global__ void bias_combine(const float* __restrict__ b0, const float* __restrict__ b1,
                             float* __restrict__ out, int n) {
  int i = blockIdx.x * blockDim.x + threadIdx.x;
  if (i < n) out[i] = b0[i] + b1[i];
}

// init recurrent state: h/c f32 copies + bf16 h copies + zero input-feed
__global__ void init_state(const float* __restrict__ h0, const float* __restrict__ c0,
                           float* __restrict__ h0s, float* __restrict__ c0s,
                           float* __restrict__ h1s, float* __restrict__ c1s,
                           bf16* __restrict__ h0bf, bf16* __restrict__ h1bf,
                           bf16* __restrict__ feed) {
  int i = blockIdx.x * blockDim.x + threadIdx.x;   // B*H
  if (i >= B * H) return;
  h0s[i] = h0[i];             c0s[i] = c0[i];
  h1s[i] = h0[B * H + i];     c1s[i] = c0[B * H + i];
  h0bf[i] = (bf16)h0s[i];
  h1bf[i] = (bf16)h1s[i];
  feed[i] = (bf16)0.f;
}

// ---------------------------------------------------------------------------
// Per-step kernels
// ---------------------------------------------------------------------------
// x = [emb_t (f32->bf16) | input_feed (bf16)]  -> (B, E+H) bf16
__global__ void build_x(const float* __restrict__ inp_t, const bf16* __restrict__ feed,
                        bf16* __restrict__ x) {
  int i = blockIdx.x * blockDim.x + threadIdx.x;   // B*(E+H)
  if (i >= B * KX) return;
  int b = i / KX, c = i % KX;
  x[i] = (c < E) ? (bf16)inp_t[(long)b * E + c] : feed[(long)b * H + (c - E)];
}

// PyTorch gate order i,f,g,o.  In-place-capable on c.
__global__ void lstm_pointwise(const float* __restrict__ gates, const float* c_in,
                               float* c_out, float* __restrict__ h_out,
                               bf16* __restrict__ h_out_bf) {
  int i = blockIdx.x * blockDim.x + threadIdx.x;   // B*H
  if (i >= B * H) return;
  int b = i >> 10, h = i & (H - 1);
  const float* g = gates + (long)b * G4H;
  float gi = g[h], gf = g[H + h], gg = g[2 * H + h], go = g[3 * H + h];
  float si = 1.f / (1.f + __expf(-gi));
  float sf = 1.f / (1.f + __expf(-gf));
  float so = 1.f / (1.f + __expf(-go));
  float c  = sf * c_in[i] + si * tanhf(gg);
  float hh = so * tanhf(c);
  c_out[i] = c;
  h_out[i] = hh;
  h_out_bf[i] = (bf16)hh;
}

// Fused attention: score -> softmax -> context vector.  One block per batch b,
// 128 threads (== S).  Writes align to attns output and cvec (B,H) f32.
__launch_bounds__(128)
__global__ void attention_fused(const float* __restrict__ h1,     // (B,H)
                                const float* __restrict__ ctx_p,  // (B,S,H)
                                const float* __restrict__ context,// (S,B,H)
                                float* __restrict__ cvec,         // (B,H)
                                float* __restrict__ attns_t)      // (B,S)
{
  __shared__ float sh_h[H];
  __shared__ float sh_al[S];
  __shared__ float red[S];
  const int b = blockIdx.x;
  const int tid = threadIdx.x;

  for (int h = tid; h < H; h += S) sh_h[h] = h1[(long)b * H + h];
  __syncthreads();

  // score_s = h1[b,:] . ctx_p[b,s,:]
  const float* cp = ctx_p + ((long)b * S + tid) * H;
  float sc = 0.f;
  for (int h = 0; h < H; ++h) sc += sh_h[h] * cp[h];

  // softmax over S (block-wide tree reductions)
  red[tid] = sc; __syncthreads();
  for (int off = S / 2; off > 0; off >>= 1) {
    if (tid < off) red[tid] = fmaxf(red[tid], red[tid + off]);
    __syncthreads();
  }
  float m = red[0]; __syncthreads();
  float e = __expf(sc - m);
  red[tid] = e; __syncthreads();
  for (int off = S / 2; off > 0; off >>= 1) {
    if (tid < off) red[tid] += red[tid + off];
    __syncthreads();
  }
  float al = e / red[0];
  sh_al[tid] = al;
  attns_t[(long)b * S + tid] = al;
  __syncthreads();

  // cvec[b,h] = sum_s align[s] * context[s,b,h]   (8 columns per thread)
#pragma unroll
  for (int hh = 0; hh < H / S; ++hh) {
    int h = hh * S + tid;
    float acc = 0.f;
    for (int s2 = 0; s2 < S; ++s2)
      acc += sh_al[s2] * context[((long)s2 * B + b) * H + h];
    cvec[(long)b * H + h] = acc;
  }
}

// cat2 = [cvec | h1] -> (B, 2H) bf16
__global__ void build_cat2(const float* __restrict__ cvec, const float* __restrict__ h1,
                           bf16* __restrict__ cat2) {
  int i = blockIdx.x * blockDim.x + threadIdx.x;   // B*2H
  if (i >= B * K2) return;
  int b = i / K2, c = i % K2;
  cat2[i] = (c < H) ? (bf16)cvec[(long)b * H + c] : (bf16)h1[(long)b * H + (c - H)];
}

// attn_h = tanh(pre);  write outs[t] (f32) + next input-feed (bf16)
__global__ void tanh_out(const float* __restrict__ pre, float* __restrict__ outs_t,
                         bf16* __restrict__ feed) {
  int i = blockIdx.x * blockDim.x + threadIdx.x;   // B*H
  if (i >= B * H) return;
  float v = tanhf(pre[i]);
  outs_t[i] = v;
  feed[i] = (bf16)v;
}

// final hF/cF copy:  d_out layout [outs | hF | cF | attns]
__global__ void copy_final(const float* __restrict__ h0s, const float* __restrict__ h1s,
                           const float* __restrict__ c0s, const float* __restrict__ c1s,
                           float* __restrict__ hF, float* __restrict__ cF) {
  int i = blockIdx.x * blockDim.x + threadIdx.x;   // B*H
  if (i >= B * H) return;
  hF[i] = h0s[i];  hF[B * H + i] = h1s[i];
  cF[i] = c0s[i];  cF[B * H + i] = c1s[i];
}

// ---------------------------------------------------------------------------
// Host-side orchestration
// ---------------------------------------------------------------------------
static inline char* ws_alloc(char*& p, size_t bytes) {
  char* r = p;
  p += (bytes + 255) & ~((size_t)255);
  return r;
}

extern "C" void kernel_launch(void* const* d_in, const int* in_sizes, int n_in,
                              void* d_out, int out_size, void* d_ws, size_t ws_size,
                              hipStream_t stream) {
  (void)in_sizes; (void)n_in; (void)out_size; (void)ws_size;

  const float* inp     = (const float*)d_in[0];   // (T,B,E)
  const float* context = (const float*)d_in[1];   // (S,B,H)
  const float* h0_in   = (const float*)d_in[2];   // (2,B,H)
  const float* c0_in   = (const float*)d_in[3];   // (2,B,H)
  const float* W_ih0   = (const float*)d_in[4];   // (4H, E+H)
  const float* b_ih0   = (const float*)d_in[5];
  const float* W_hh0   = (const float*)d_in[6];   // (4H, H)
  const float* b_hh0   = (const float*)d_in[7];
  const float* W_ih1   = (const float*)d_in[8];   // (4H, H)
  const float* b_ih1   = (const float*)d_in[9];
  const float* W_hh1   = (const float*)d_in[10];  // (4H, H)
  const float* b_hh1   = (const float*)d_in[11];
  const float* W_a     = (const float*)d_in[12];  // (H, H)
  const float* W_out   = (const float*)d_in[13];  // (H, 2H)

  float* out = (float*)d_out;
  float* outs_base  = out;                               // T*B*H
  float* hF_base    = out + (long)T * B * H;             // 2*B*H
  float* cF_base    = hF_base + 2L * B * H;              // 2*B*H
  float* attns_base = cF_base + 2L * B * H;              // T*B*S

  // workspace carve-up
  char* p = (char*)d_ws;
  bf16* Wih0bf = (bf16*)ws_alloc(p, (size_t)G4H * KX * 2);
  bf16* Whh0bf = (bf16*)ws_alloc(p, (size_t)G4H * H * 2);
  bf16* Wih1bf = (bf16*)ws_alloc(p, (size_t)G4H * H * 2);
  bf16* Whh1bf = (bf16*)ws_alloc(p, (size_t)G4H * H * 2);
  bf16* WaTbf  = (bf16*)ws_alloc(p, (size_t)H * H * 2);
  bf16* Woutbf = (bf16*)ws_alloc(p, (size_t)H * K2 * 2);
  bf16* ctxbt  = (bf16*)ws_alloc(p, (size_t)B * S * H * 2);
  float* ctx_p = (float*)ws_alloc(p, (size_t)B * S * H * 4);
  float* gates0 = (float*)ws_alloc(p, (size_t)B * G4H * 4);
  float* gates1 = (float*)ws_alloc(p, (size_t)B * G4H * 4);
  float* h0s = (float*)ws_alloc(p, (size_t)B * H * 4);
  float* c0s = (float*)ws_alloc(p, (size_t)B * H * 4);
  float* h1s = (float*)ws_alloc(p, (size_t)B * H * 4);
  float* c1s = (float*)ws_alloc(p, (size_t)B * H * 4);
  bf16* h0bf = (bf16*)ws_alloc(p, (size_t)B * H * 2);
  bf16* h1bf = (bf16*)ws_alloc(p, (size_t)B * H * 2);
  bf16* feed = (bf16*)ws_alloc(p, (size_t)B * H * 2);
  bf16* xbf  = (bf16*)ws_alloc(p, (size_t)B * KX * 2);
  bf16* cat2 = (bf16*)ws_alloc(p, (size_t)B * K2 * 2);
  float* cvec = (float*)ws_alloc(p, (size_t)B * H * 4);
  float* pre  = (float*)ws_alloc(p, (size_t)B * H * 4);
  float* bias0 = (float*)ws_alloc(p, (size_t)G4H * 4);
  float* bias1 = (float*)ws_alloc(p, (size_t)G4H * 4);

  auto blocks = [](long n, int bs) { return (unsigned)((n + bs - 1) / bs); };

  // ---- one-time setup: bf16 weight conversion, ctx transpose, ctx_p GEMM ----
  cvt_f32_bf16<<<blocks((long)G4H * KX, 256), 256, 0, stream>>>(W_ih0, Wih0bf, G4H * KX);
  cvt_f32_bf16<<<blocks((long)G4H * H, 256), 256, 0, stream>>>(W_hh0, Whh0bf, G4H * H);
  cvt_f32_bf16<<<blocks((long)G4H * H, 256), 256, 0, stream>>>(W_ih1, Wih1bf, G4H * H);
  cvt_f32_bf16<<<blocks((long)G4H * H, 256), 256, 0, stream>>>(W_hh1, Whh1bf, G4H * H);
  cvt_f32_bf16<<<blocks((long)H * K2, 256), 256, 0, stream>>>(W_out, Woutbf, H * K2);
  cvt_f32_bf16_T<<<blocks((long)H * H, 256), 256, 0, stream>>>(W_a, WaTbf, H, H);
  ctx_transpose_bf16<<<blocks((long)S * B * H, 256), 256, 0, stream>>>(context, ctxbt);
  bias_combine<<<blocks(G4H, 256), 256, 0, stream>>>(b_ih0, b_hh0, bias0, G4H);
  bias_combine<<<blocks(G4H, 256), 256, 0, stream>>>(b_ih1, b_hh1, bias1, G4H);
  init_state<<<blocks((long)B * H, 256), 256, 0, stream>>>(h0_in, c0_in, h0s, c0s, h1s, c1s,
                                                           h0bf, h1bf, feed);
  // ctx_p = ctx_bt @ W_a   (M = B*S = 8192, N = H, K = H); Bw = W_a^T (N,K)
  gemm_wmma_bf16<<<dim3(H / 64, (B * S) / 16), 32, 0, stream>>>(
      ctxbt, H, WaTbf, H, ctx_p, H, H, nullptr, 0);

  // ---- recurrence ----
  for (int t = 0; t < T; ++t) {
    const float* inp_t = inp + (long)t * B * E;
    float* outs_t  = outs_base  + (long)t * B * H;
    float* attns_t = attns_base + (long)t * B * S;

    build_x<<<blocks((long)B * KX, 256), 256, 0, stream>>>(inp_t, feed, xbf);

    // layer-0 gates = x @ W_ih0^T + bias ; += h0 @ W_hh0^T
    gemm_wmma_bf16<<<dim3(G4H / 64, B / 16), 32, 0, stream>>>(
        xbf, KX, Wih0bf, KX, gates0, G4H, KX, bias0, 0);
    gemm_wmma_bf16<<<dim3(G4H / 64, B / 16), 32, 0, stream>>>(
        h0bf, H, Whh0bf, H, gates0, G4H, H, nullptr, 1);
    lstm_pointwise<<<blocks((long)B * H, 256), 256, 0, stream>>>(
        gates0, c0s, c0s, h0s, h0bf);

    // layer-1 gates = h0n @ W_ih1^T + bias ; += h1 @ W_hh1^T
    gemm_wmma_bf16<<<dim3(G4H / 64, B / 16), 32, 0, stream>>>(
        h0bf, H, Wih1bf, H, gates1, G4H, H, bias1, 0);
    gemm_wmma_bf16<<<dim3(G4H / 64, B / 16), 32, 0, stream>>>(
        h1bf, H, Whh1bf, H, gates1, G4H, H, nullptr, 1);
    lstm_pointwise<<<blocks((long)B * H, 256), 256, 0, stream>>>(
        gates1, c1s, c1s, h1s, h1bf);

    // attention (score/softmax/context) + output projection + tanh
    attention_fused<<<B, S, 0, stream>>>(h1s, ctx_p, context, cvec, attns_t);
    build_cat2<<<blocks((long)B * K2, 256), 256, 0, stream>>>(cvec, h1s, cat2);
    gemm_wmma_bf16<<<dim3(H / 64, B / 16), 32, 0, stream>>>(
        cat2, K2, Woutbf, K2, pre, H, K2, nullptr, 0);
    tanh_out<<<blocks((long)B * H, 256), 256, 0, stream>>>(pre, outs_t, feed);
  }

  copy_final<<<blocks((long)B * H, 256), 256, 0, stream>>>(h0s, h1s, c0s, c1s,
                                                           hF_base, cF_base);
}
